// HierarchicalInterventionVAE_1133871366527
// MI455X (gfx1250) — matrix operand order repaired and checked by reading, not compiled, not run on hardware
//
#include <hip/hip_runtime.h>
#include <hip/hip_bf16.h>

typedef __attribute__((ext_vector_type(16))) _Float16 v16h;
typedef __attribute__((ext_vector_type(8)))  float    v8f;
typedef __attribute__((ext_vector_type(4)))  float    v4f;

namespace {

constexpr int kB = 8192;
constexpr int kD = 128;
constexpr int kIters = 40;

// ---- explicit address-space accessors (global_load / ds_* instead of flat) ----
__device__ __forceinline__ float ld_g(const float* p) {
  return *(const __attribute__((address_space(1))) float*)p;
}
__device__ __forceinline__ v4f ld_g4(const float* p) {
  return *(const __attribute__((address_space(1))) v4f*)p;
}
__device__ __forceinline__ void st_g(float* p, float v) {
  *(__attribute__((address_space(1))) float*)p = v;
}
__device__ __forceinline__ float ld_s(const float* p) {
  return *(const __attribute__((address_space(3))) float*)p;
}
__device__ __forceinline__ void st_s(float* p, float v) {
  *(__attribute__((address_space(3))) float*)p = v;
}
__device__ __forceinline__ void st_sh(_Float16* p, _Float16 v) {
  *(__attribute__((address_space(3))) _Float16*)p = v;
}
__device__ __forceinline__ v16h ld_s_frag(const _Float16* p) {
  return *(const __attribute__((address_space(3))) v16h*)p;
}

struct Params {
  const float *pf, *X, *eps_base, *eps_treat, *eps_t1, *eps_t1_cf;
  const float *beW0,*beB0,*beW1,*beB1,*beW2,*beB2;   // base_enc  [128,64,64,128]
  const float *teW0,*teB0,*teW1,*teB1,*teW2,*teB2;   // treat_enc [65,64,64,128]
  const float *cW0,*cB0,*cW1,*cB1;                   // combine   [128,64,128]
  const float *gW0,*gB0,*gW1,*gB1;                   // proto_gate[1,32,64]
  const float *prW0,*prB0,*prW1,*prB1;               // prior_t1  [128,64,256]
  const float *dW0,*dB0,*dW1,*dB1,*dW2,*dB2;         // decoder   [128,64,64,128]
  const float *protos;                               // (64,128)
  float *recon,*recon_cf,*klb,*klt,*kl1,*wcost,*wcost_cf;
};

// Stage W (Kact x N logical, element (k,n) at W[k*wRS + n*wCS]) into LDS f16,
// laid out so each lane's 16-element B fragment is contiguous (32B):
//   off = (((n/16)*KT + k/32)*32 + lane(k%32,n%16))*16 + e
// Rows k in [Kact, Kdim) are zero (zero-fill pass). N in {64,128}.
__device__ __forceinline__ void stage_weights(
    const float* __restrict__ W, int wRS, int wCS, int Kdim, int Kact, int N,
    const float* __restrict__ bias, _Float16* s_wt, float* s_bias)
{
  const int lane = threadIdx.x;
  const int KT = Kdim >> 5;
  const int nsh = (N == 128) ? 7 : 6;
  if (Kact < Kdim) {
    for (int i = lane; i < Kdim * N; i += 32) st_sh(&s_wt[i], (_Float16)0.0f);
  }
  for (int i = lane; i < Kact * N; i += 32) {
    int k = i >> nsh, n = i & (N - 1);
    int kt = k >> 5, kr = k & 31, nt = n >> 4, nc = n & 15;
    int fl = ((kr >> 4) << 4) + nc;     // lane owning this element
    int e  = kr & 15;                   // element slot within fragment
    int off = (((nt * KT) + kt) * 32 + fl) * 16 + e;
    st_sh(&s_wt[off], (_Float16)ld_g(&W[k * wRS + n * wCS]));
  }
  if (bias) {
    for (int i = lane; i < N; i += 32) st_s(&s_bias[i], ld_g(&bias[i]));
  }
}

// Y[16 x N] = act( X[16 x Kdim] @ W + bias ); X in LDS (row-major ld=ldin, f32),
// W pre-staged as f16 fragments in s_wt. A-fragments hoisted across N-tiles.
__device__ __forceinline__ void linear_staged(
    const float* __restrict__ in, int ldin, int Kdim,
    const _Float16* __restrict__ s_wt, const float* __restrict__ s_bias,
    bool bias_en, int N, float* __restrict__ out, int ldout,
    bool relu, bool to_global)
{
  const int lane = threadIdx.x;
  const int m    = lane & 15;
  const int khiA = (lane >> 4) << 3;
  const int nB   = lane & 15;
  const int m0   = (lane >> 4) << 3;
  const int KT   = Kdim >> 5;

  v16h afrag[4];                        // A 16x32 f16 fragments, all K tiles
#pragma unroll
  for (int kt = 0; kt < 4; ++kt) {
    if (kt < KT) {
#pragma unroll
      for (int e = 0; e < 16; ++e) {    // A layout: k = 16*(lane/16) + e (matches staged B)
        int k = (kt << 5) + (khiA << 1) + e;
        afrag[kt][e] = (_Float16)ld_s(in + m * ldin + k);
      }
    }
  }
#pragma unroll
  for (int nt = 0; nt < (N >> 4); ++nt) {
    v8f acc = {};
#pragma unroll
    for (int kt = 0; kt < 4; ++kt) {
      if (kt < KT) {
        v16h b = ld_s_frag(s_wt + (((nt * KT) + kt) * 32 + lane) * 16);
        acc = __builtin_amdgcn_wmma_f32_16x16x32_f16(false, afrag[kt], false, b,
                                                     (short)0, acc, false, false);
      }
    }
#pragma unroll
    for (int r = 0; r < 8; ++r) {
      int col = (nt << 4) + nB;
      float v = acc[r];
      if (bias_en) v += ld_s(s_bias + col);
      if (relu) v = fmaxf(v, 0.0f);
      if (to_global) st_g(out + (size_t)(m0 + r) * ldout + col, v);
      else           st_s(out + (m0 + r) * ldout + col, v);
    }
  }
}

// z = mu + eps*exp(lv/2), KL row-sum; mu|lv packed in s_x (16x128). Lane pair
// (l, l^16) shares row l&15; columns split 32/32; reduce with shfl_xor(16).
__device__ __forceinline__ void reparam64(const float* s_x, const float* epsg,
                                          int row0, float* s_z, float* klout)
{
  const int lane = threadIdx.x, mm = lane & 15, h = lane >> 4;
  const int g = row0 + mm, j0 = h * 32;
  float kl = 0.f;
  for (int j = j0; j < j0 + 32; ++j) {
    float mu = ld_s(s_x + mm * 128 + j);
    float lv = ld_s(s_x + mm * 128 + 64 + j);
    float e  = ld_g(epsg + (size_t)g * 64 + j);
    st_s(s_z + mm * 64 + j, mu + e * __expf(0.5f * lv));
    kl += 1.0f + lv - mu * mu - __expf(lv);
  }
  kl += __shfl_xor(kl, 16, 32);
  if (h == 0) st_g(klout + g, -0.5f * kl);
}

// theta0 (16x128) in s_c -> comb weights s_w, t0v in s_a, sum(w*cost) out.
__device__ void proto_block(const Params& p, int row0, bool use_pf,
                            float* s_a, float* s_b, float* s_c,
                            float* s_q, float* s_w,
                            float* s_pn, float* s_tn, float* s_pf,
                            _Float16* s_wt, float* s_bias, float* out_wcost)
{
  const int lane = threadIdx.x, mm = lane & 15, h = lane >> 4;
  {                                             // ||theta0_m||^2
    float s = 0.f;
    for (int j = h * 64; j < h * 64 + 64; ++j) {
      float v = ld_s(s_c + mm * 128 + j); s += v * v;
    }
    s += __shfl_xor(s, 16, 32);
    if (h == 0) st_s(s_tn + mm, s);
  }
  __syncthreads();
  // dot = theta0 @ protos^T : (16x128)@(128x64)
  stage_weights(p.protos, 1, 128, 128, 128, 64, nullptr, s_wt, s_bias);
  __syncthreads();
  linear_staged(s_c, 128, 128, s_wt, s_bias, false, 64, s_b, 64, false, false);
  __syncthreads();
  for (int idx = lane; idx < 16 * 64; idx += 32) {   // cost, q = exp(-cost/eps)
    int r = idx >> 6, k = idx & 63;
    float c = ld_s(s_tn + r) + ld_s(s_pn + k) - 2.0f * ld_s(s_b + idx);
    st_s(s_a + idx, c);
    st_s(s_q + idx, __expf(-c * 20.0f));             // 1/0.05
  }
  __syncthreads();
  // Collapsed Sinkhorn (M[b,k,l]=cost[b,l]): q,v in registers, shfl reduce.
  {
    const int base = mm * 64 + h * 32;
    const float invK = 1.0f / 64.0f;
    float q[32], v[32];
#pragma unroll
    for (int l = 0; l < 32; ++l) { q[l] = ld_s(s_q + base + l); v[l] = invK; }
    float u = invK;
#pragma unroll 1
    for (int it = 0; it < kIters; ++it) {
      float s = 0.f;
#pragma unroll
      for (int l = 0; l < 32; ++l) s += q[l] * v[l];
      s += __shfl_xor(s, 16, 32);
      u = invK / (s + 1e-9f);
      float ku = 64.0f * u;
#pragma unroll
      for (int l = 0; l < 32; ++l) v[l] = invK / (q[l] * ku + 1e-9f);
    }
    float uk = 64.0f * u;
#pragma unroll
    for (int l = 0; l < 32; ++l) st_s(s_b + base + l, uk * q[l] * v[l]);
  }
  __syncthreads();
  for (int idx = lane; idx < 16 * 32; idx += 32) {   // gate hidden (reuse s_q)
    int r = idx >> 5, hh = idx & 31;
    float x = use_pf ? ld_s(s_pf + r) : 0.0f;
    st_s(s_q + idx, fmaxf(x * ld_g(p.gW0 + hh) + ld_g(p.gB0 + hh), 0.0f));
  }
  __syncthreads();
  stage_weights(p.gW1, 64, 1, 32, 32, 64, p.gB1, s_wt, s_bias);
  __syncthreads();
  linear_staged(s_q, 32, 32, s_wt, s_bias, true, 64, s_w, 64, false, false);
  __syncthreads();
  {                                             // softmax gate, comb, sum(w*cost)
    const int base = mm * 64 + h * 32;
    float lg[32], c[32];
#pragma unroll
    for (int l = 0; l < 32; ++l) lg[l] = ld_s(s_w + base + l);
    float mx = -3.4e38f;
#pragma unroll
    for (int l = 0; l < 32; ++l) mx = fmaxf(mx, lg[l]);
    mx = fmaxf(mx, __shfl_xor(mx, 16, 32));
    float sum = 0.f;
#pragma unroll
    for (int l = 0; l < 32; ++l) { lg[l] = __expf(lg[l] - mx); sum += lg[l]; }
    sum += __shfl_xor(sum, 16, 32);
    float inv_sum = 1.0f / sum;
    float cs = 0.f;
#pragma unroll
    for (int l = 0; l < 32; ++l) {
      c[l] = ld_s(s_b + base + l) * lg[l] * inv_sum; cs += c[l];
    }
    cs += __shfl_xor(cs, 16, 32);
    float inv = 1.0f / (cs + 1e-9f);
    float wc = 0.f;
#pragma unroll
    for (int l = 0; l < 32; ++l) {
      float cc = c[l] * inv;
      st_s(s_w + base + l, cc);
      wc += cc * ld_s(s_a + base + l);
    }
    wc += __shfl_xor(wc, 16, 32);
    if (h == 0) st_g(out_wcost + row0 + mm, wc);
  }
  __syncthreads();
  // t0v = comb @ protos : (16x64)@(64x128)
  stage_weights(p.protos, 128, 1, 64, 64, 128, nullptr, s_wt, s_bias);
  __syncthreads();
  linear_staged(s_w, 64, 64, s_wt, s_bias, false, 128, s_a, 128, false, false);
  __syncthreads();
}

// t0v (16x128) in s_a -> recon rows to global; KL1 optional.
__device__ void decode_block(const Params& p, int row0,
                             float* s_a, float* s_b, float* s_c,
                             _Float16* s_wt, float* s_bias,
                             const float* epsg, float* recong, float* kl1g)
{
  const int lane = threadIdx.x, mm = lane & 15, h = lane >> 4;
  stage_weights(p.prW0, 64, 1, 128, 128, 64, p.prB0, s_wt, s_bias);
  __syncthreads();
  linear_staged(s_a, 128, 128, s_wt, s_bias, true, 64, s_b, 64, true, false);
  __syncthreads();
  stage_weights(p.prW1, 256, 1, 64, 64, 128, p.prB1, s_wt, s_bias);        // mu
  __syncthreads();
  linear_staged(s_b, 64, 64, s_wt, s_bias, true, 128, s_c, 128, false, false);
  __syncthreads();
  stage_weights(p.prW1 + 128, 256, 1, 64, 64, 128, p.prB1 + 128, s_wt, s_bias); // lv
  __syncthreads();
  linear_staged(s_b, 64, 64, s_wt, s_bias, true, 128, s_a, 128, false, false);
  __syncthreads();
  {                                       // th1 = mu + eps*exp(lv/2); KL1
    const int g = row0 + mm; float kl = 0.f;
    for (int j = h * 64; j < h * 64 + 64; ++j) {
      float mu = ld_s(s_c + mm * 128 + j), lv = ld_s(s_a + mm * 128 + j);
      float e  = ld_g(epsg + (size_t)g * 128 + j);
      st_s(s_c + mm * 128 + j, mu + e * __expf(0.5f * lv));
      kl += 1.0f + lv - mu * mu - __expf(lv);
    }
    kl += __shfl_xor(kl, 16, 32);
    if (kl1g && h == 0) st_g(kl1g + g, -0.5f * kl);
  }
  __syncthreads();
  stage_weights(p.dW0, 64, 1, 128, 128, 64, p.dB0, s_wt, s_bias);
  __syncthreads();
  linear_staged(s_c, 128, 128, s_wt, s_bias, true, 64, s_b, 64, true, false);
  __syncthreads();
  stage_weights(p.dW1, 64, 1, 64, 64, 64, p.dB1, s_wt, s_bias);
  __syncthreads();
  linear_staged(s_b, 64, 64, s_wt, s_bias, true, 64, s_a, 64, true, false);
  __syncthreads();
  stage_weights(p.dW2, 128, 1, 64, 64, 128, p.dB2, s_wt, s_bias);
  __syncthreads();
  linear_staged(s_a, 64, 64, s_wt, s_bias, true, 128,
                recong + (size_t)row0 * 128, 128, false, true);
  __syncthreads();
}

__global__ __launch_bounds__(32)
void hivae_fused_wmma_kernel(Params p)
{
  __shared__ float s_a[16 * 128];
  __shared__ float s_b[16 * 128];
  __shared__ float s_c[16 * 128];
  __shared__ float s_zb[16 * 64];
  __shared__ float s_zt[16 * 64];
  __shared__ float s_q [16 * 64];
  __shared__ float s_w [16 * 64];
  __shared__ __align__(32) _Float16 s_wt[128 * 64];   // staged weight fragments
  __shared__ float s_bias[128];
  __shared__ float s_pn[64];
  __shared__ float s_tn[16];
  __shared__ float s_pf[16];

  const int lane = threadIdx.x;
  const int row0 = blockIdx.x * 16;

  if (lane == 0) {                        // gfx1250 global_prefetch path
    __builtin_prefetch(p.beW0, 0, 3);
    __builtin_prefetch(p.protos, 0, 3);
    __builtin_prefetch(p.prW1, 0, 3);
    __builtin_prefetch(p.dW0, 0, 3);
  }

  if (lane < 16) st_s(s_pf + lane, ld_g(p.pf + row0 + lane));
  for (int j = lane; j < 64; j += 32) {   // prototype squared norms (vec4)
    float s = 0.f;
    for (int d = 0; d < 128; d += 4) {
      v4f v = ld_g4(p.protos + j * 128 + d);
      s += v.x * v.x + v.y * v.y + v.z * v.z + v.w * v.w;
    }
    st_s(s_pn + j, s);
  }
  for (int i = lane; i < 16 * 128 / 4; i += 32) {     // X tile (vec4 coalesced)
    v4f v = ld_g4(p.X + (size_t)row0 * 128 + i * 4);
    st_s(s_a + i * 4 + 0, v.x); st_s(s_a + i * 4 + 1, v.y);
    st_s(s_a + i * 4 + 2, v.z); st_s(s_a + i * 4 + 3, v.w);
  }
  __syncthreads();

  // ---- base encoder: [128 ->64 relu ->64 relu ->128] ----
  stage_weights(p.beW0, 64, 1, 128, 128, 64, p.beB0, s_wt, s_bias);
  __syncthreads();
  linear_staged(s_a, 128, 128, s_wt, s_bias, true, 64, s_b, 64, true, false);
  __syncthreads();
  stage_weights(p.beW1, 64, 1, 64, 64, 64, p.beB1, s_wt, s_bias);
  __syncthreads();
  linear_staged(s_b, 64, 64, s_wt, s_bias, true, 64, s_c, 64, true, false);
  __syncthreads();
  stage_weights(p.beW2, 128, 1, 64, 64, 128, p.beB2, s_wt, s_bias);
  __syncthreads();
  linear_staged(s_c, 64, 64, s_wt, s_bias, true, 128, s_a, 128, false, false);
  __syncthreads();
  reparam64(s_a, p.eps_base, row0, s_zb, p.klb);
  __syncthreads();

  // ---- treat encoder: input [pf | z_base | 0-pad] 16x96, Kact=65 ----
  for (int idx = lane; idx < 16 * 96; idx += 32) {
    int mm = idx / 96, c = idx - mm * 96;
    float v = 0.f;
    if (c == 0) v = ld_s(s_pf + mm);
    else if (c <= 64) v = ld_s(s_zb + mm * 64 + (c - 1));
    st_s(s_b + idx, v);
  }
  __syncthreads();
  stage_weights(p.teW0, 64, 1, 96, 65, 64, p.teB0, s_wt, s_bias);  // zero-padded K
  __syncthreads();
  linear_staged(s_b, 96, 96, s_wt, s_bias, true, 64, s_c, 64, true, false);
  __syncthreads();
  stage_weights(p.teW1, 64, 1, 64, 64, 64, p.teB1, s_wt, s_bias);
  __syncthreads();
  linear_staged(s_c, 64, 64, s_wt, s_bias, true, 64, s_b, 64, true, false);
  __syncthreads();
  stage_weights(p.teW2, 128, 1, 64, 64, 128, p.teB2, s_wt, s_bias);
  __syncthreads();
  linear_staged(s_b, 64, 64, s_wt, s_bias, true, 128, s_a, 128, false, false);
  __syncthreads();
  reparam64(s_a, p.eps_treat, row0, s_zt, p.klt);
  __syncthreads();

  for (int pass = 0; pass < 2; ++pass) {
    const bool factual = (pass == 0);
    // combine input: [z_base | z_treat or 0]
    for (int idx = lane; idx < 16 * 128; idx += 32) {
      int mm = idx >> 7, c = idx & 127;
      float v = (c < 64) ? ld_s(s_zb + mm * 64 + c)
                         : (factual ? ld_s(s_zt + mm * 64 + (c - 64)) : 0.0f);
      st_s(s_a + idx, v);
    }
    __syncthreads();
    stage_weights(p.cW0, 64, 1, 128, 128, 64, p.cB0, s_wt, s_bias);
    __syncthreads();
    linear_staged(s_a, 128, 128, s_wt, s_bias, true, 64, s_b, 64, true, false);
    __syncthreads();
    stage_weights(p.cW1, 128, 1, 64, 64, 128, p.cB1, s_wt, s_bias);
    __syncthreads();
    linear_staged(s_b, 64, 64, s_wt, s_bias, true, 128, s_c, 128, false, false);
    __syncthreads();
    proto_block(p, row0, /*use_pf=*/factual, s_a, s_b, s_c, s_q, s_w,
                s_pn, s_tn, s_pf, s_wt, s_bias,
                factual ? p.wcost : p.wcost_cf);
    decode_block(p, row0, s_a, s_b, s_c, s_wt, s_bias,
                 factual ? p.eps_t1 : p.eps_t1_cf,
                 factual ? p.recon : p.recon_cf,
                 factual ? p.kl1 : nullptr);
  }
}

} // namespace

extern "C" void kernel_launch(void* const* d_in, const int* in_sizes, int n_in,
                              void* d_out, int out_size, void* d_ws, size_t ws_size,
                              hipStream_t stream) {
  (void)in_sizes; (void)n_in; (void)out_size; (void)d_ws; (void)ws_size;
  Params p;
  int i = 0;
  p.pf        = (const float*)d_in[i++];
  p.X         = (const float*)d_in[i++];
  p.eps_base  = (const float*)d_in[i++];
  p.eps_treat = (const float*)d_in[i++];
  p.eps_t1    = (const float*)d_in[i++];
  p.eps_t1_cf = (const float*)d_in[i++];
  p.beW0=(const float*)d_in[i++]; p.beB0=(const float*)d_in[i++];
  p.beW1=(const float*)d_in[i++]; p.beB1=(const float*)d_in[i++];
  p.beW2=(const float*)d_in[i++]; p.beB2=(const float*)d_in[i++];
  p.teW0=(const float*)d_in[i++]; p.teB0=(const float*)d_in[i++];
  p.teW1=(const float*)d_in[i++]; p.teB1=(const float*)d_in[i++];
  p.teW2=(const float*)d_in[i++]; p.teB2=(const float*)d_in[i++];
  p.cW0 =(const float*)d_in[i++]; p.cB0 =(const float*)d_in[i++];
  p.cW1 =(const float*)d_in[i++]; p.cB1 =(const float*)d_in[i++];
  p.gW0 =(const float*)d_in[i++]; p.gB0 =(const float*)d_in[i++];
  p.gW1 =(const float*)d_in[i++]; p.gB1 =(const float*)d_in[i++];
  p.prW0=(const float*)d_in[i++]; p.prB0=(const float*)d_in[i++];
  p.prW1=(const float*)d_in[i++]; p.prB1=(const float*)d_in[i++];
  p.dW0 =(const float*)d_in[i++]; p.dB0 =(const float*)d_in[i++];
  p.dW1 =(const float*)d_in[i++]; p.dB1 =(const float*)d_in[i++];
  p.dW2 =(const float*)d_in[i++]; p.dB2 =(const float*)d_in[i++];
  p.protos = (const float*)d_in[i++];

  float* out = (float*)d_out;
  p.recon    = out;
  p.recon_cf = out + (size_t)kB * kD;
  p.klb      = out + 2ull * kB * kD;
  p.klt      = p.klb + kB;
  p.kl1      = p.klt + kB;
  p.wcost    = p.kl1 + kB;
  p.wcost_cf = p.wcost + kB;

  hivae_fused_wmma_kernel<<<kB / 16, 32, 0, stream>>>(p);
}
